// MOERouter_8873402433830
// MI455X (gfx1250) — matrix operand order repaired and checked by reading, not compiled, not run on hardware
//
#include <hip/hip_runtime.h>
#include <hip/hip_bf16.h>

// ---------------- types for WMMA ----------------
typedef __attribute__((ext_vector_type(16))) __bf16 v16bf;
typedef __attribute__((ext_vector_type(8)))  __bf16 v8bf;
typedef __attribute__((ext_vector_type(8)))  float  v8f;

__device__ __forceinline__ v16bf cat8(v8bf lo, v8bf hi) {
    return __builtin_shufflevector(lo, hi, 0,1,2,3,4,5,6,7,8,9,10,11,12,13,14,15);
}

__device__ __forceinline__ float gelu_erf(float v) {
    return 0.5f * v * (1.0f + erff(v * 0.70710678118654752f));
}

#define B_SZ   512
#define C_IN   32
#define L_SZ   512
#define E_SZ   8
#define POOL   16
#define D_IN   (C_IN * L_SZ)     // 16384
#define D_OUT  768

// ---------------- kernel 1: zero output (+ counts, defensively) ----------------
__global__ void zero_kernel(float* __restrict__ out, int* __restrict__ counts) {
    int i = blockIdx.x * blockDim.x + threadIdx.x;
    if (i < B_SZ * D_OUT) out[i] = 0.0f;
    if (i < E_SZ) counts[i] = 0;
}

// ---------------- kernel 2: router (conv+gelu+pool+bn+fc1+gelu+fc2+gumbel+top2) ----------------
__global__ __launch_bounds__(256) void router_kernel(
    const float* __restrict__ x,      const float* __restrict__ gu,
    const float* __restrict__ conv_w, const float* __restrict__ conv_b,
    const float* __restrict__ bng,    const float* __restrict__ bnb,
    const float* __restrict__ bnm,    const float* __restrict__ bnv,
    const float* __restrict__ fc1w,   const float* __restrict__ fc1b,
    const float* __restrict__ fc2w,   const float* __restrict__ fc2b,
    int* __restrict__ topi, float* __restrict__ wts)
{
    __shared__ float wc[64 * 32 * 3];     // 24 KB conv weights
    __shared__ float xs[32 * 34];         // input slab (34 = 32 + halo)
    __shared__ float pooled[64 * POOL];   // 1024 post-BN features
    __shared__ float h1[128];
    __shared__ float lg[8];

    const int b   = blockIdx.x;
    const int tid = threadIdx.x;
    const float* xb = x + (size_t)b * C_IN * L_SZ;

    for (int i = tid; i < 64 * 32 * 3; i += 256) wc[i] = conv_w[i];

    for (int j = 0; j < POOL; ++j) {
        __syncthreads();  // xs reuse + (first iter) wc ready
        for (int i = tid; i < 32 * 34; i += 256) {
            int ci = i / 34, ll = i % 34;
            int l = j * 32 - 1 + ll;
            xs[i] = (l >= 0 && l < L_SZ) ? xb[ci * L_SZ + l] : 0.0f;
        }
        __syncthreads();
        if (tid < 64) {
            const int o = tid;
            float s = 0.0f;
            for (int ll = 0; ll < 32; ++ll) {
                float acc = conv_b[o];
                #pragma unroll 4
                for (int ci = 0; ci < 32; ++ci) {
                    const float* wp = &wc[(o * 32 + ci) * 3];
                    const float* xp = &xs[ci * 34 + ll];      // xp[k] == x[l-1+k]
                    acc += xp[0] * wp[0] + xp[1] * wp[1] + xp[2] * wp[2];
                }
                s += gelu_erf(acc);
            }
            float p = s * (1.0f / 32.0f);
            const int idx = o * POOL + j;
            p = (p - bnm[idx]) * rsqrtf(bnv[idx] + 1e-5f) * bng[idx] + bnb[idx];
            pooled[idx] = p;
        }
    }
    __syncthreads();

    if (tid < 128) {
        float acc = fc1b[tid];
        const float* wr = fc1w + tid * 1024;
        for (int i = 0; i < 1024; ++i) acc += wr[i] * pooled[i];
        h1[tid] = gelu_erf(acc);
    }
    __syncthreads();

    if (tid < E_SZ) {
        float acc = fc2b[tid];
        const float* wr = fc2w + tid * 128;
        for (int i = 0; i < 128; ++i) acc += wr[i] * h1[i];
        float u = gu[b * E_SZ + tid];
        lg[tid] = acc + (-logf(-logf(u)));   // gumbel noise, tau=1
    }
    __syncthreads();

    if (tid == 0) {
        float mx = lg[0];
        for (int i = 1; i < E_SZ; ++i) mx = fmaxf(mx, lg[i]);
        float p[E_SZ]; float sum = 0.0f;
        for (int i = 0; i < E_SZ; ++i) { p[i] = expf(lg[i] - mx); sum += p[i]; }
        float inv = 1.0f / sum;
        for (int i = 0; i < E_SZ; ++i) p[i] *= inv;
        int i0 = 0;
        for (int i = 1; i < E_SZ; ++i) if (p[i] > p[i0]) i0 = i;   // strict > : lowest idx on ties
        int i1 = (i0 == 0) ? 1 : 0;
        for (int i = 0; i < E_SZ; ++i) { if (i == i0) continue; if (p[i] > p[i1]) i1 = i; }
        float v0 = p[i0], v1 = p[i1];
        float winv = 1.0f / (v0 + v1 + 1e-8f);
        topi[b * 2 + 0] = i0;  topi[b * 2 + 1] = i1;
        wts [b * 2 + 0] = v0 * winv;  wts[b * 2 + 1] = v1 * winv;
    }
}

// ---------------- kernel 3: deterministic dispatch (row lists per expert) ----------------
__global__ void dispatch_kernel(const int* __restrict__ topi, const float* __restrict__ wts,
                                int* __restrict__ counts, int* __restrict__ rows,
                                float* __restrict__ rwt)
{
    int e = threadIdx.x;
    if (e >= E_SZ) return;
    int c = 0;
    for (int b = 0; b < B_SZ; ++b) {
        #pragma unroll
        for (int s = 0; s < 2; ++s) {
            if (topi[b * 2 + s] == e) {
                rows[e * B_SZ + c] = b;
                rwt [e * B_SZ + c] = wts[b * 2 + s];
                ++c;
            }
        }
    }
    counts[e] = c;
}

// ---------------- kernel 4: gathered expert GEMM, bf16 WMMA, reg-pipelined staging ----------
// Block: 64 gathered rows x 128 cols of one expert; 256 threads = 8 wave32.
// Per chunk (K=128): convert+store pre-loaded regs -> LDS, barrier, issue next
// chunk's global_load_b128s (in flight during compute), 16 WMMAs, barrier.
#define MT    64
#define NT    128
#define KC    128
#define KPAD  136   // bf16 elems; 272B row stride (16B aligned, rotates LDS banks)
#define AREGS ((MT * KC) / 4 / 256)   // 8  float4 per thread
#define BREGS ((NT * KC) / 4 / 256)   // 16 float4 per thread

__global__ __launch_bounds__(256) void expert_kernel(
    const float* __restrict__ x,   const float* __restrict__ ew,
    const float* __restrict__ ebias,
    const int* __restrict__ counts, const int* __restrict__ rows,
    const float* __restrict__ rwt,  float* __restrict__ out)
{
    __shared__ __align__(16) __bf16 As[MT * KC];    // 16 KB, row-major [m][k]
    __shared__ __align__(16) __bf16 Bs[NT * KPAD];  // 34 KB, transposed [n][k]
    __shared__ int   rl[MT];
    __shared__ float wl[MT];

    const int e  = blockIdx.x >> 3;          // 8 m-tiles per expert
    const int mt = blockIdx.x & 7;
    const int n0 = blockIdx.y * NT;
    const int cnt = counts[e];
    if (mt * MT >= cnt) return;

    const int tid = threadIdx.x;
    for (int i = tid; i < MT; i += 256) {
        int r = mt * MT + i;
        if (r < cnt) { rl[i] = rows[e * B_SZ + r]; wl[i] = rwt[e * B_SZ + r]; }
        else         { rl[i] = 0;                  wl[i] = 0.0f; }
    }
    __syncthreads();

    const float* W = ew + (size_t)e * D_IN * D_OUT;
    const unsigned lane = tid & 31u;
    const unsigned wave = tid >> 5;          // 0..7 -> 16-col stripe
    const int nn = lane & 15;                // N (and A-row) index within tile
    const int lh = lane >> 4;                // lane-half selects K sub-offset
    const int col = wave * 16 + nn;

    v8f acc0 = {}, acc1 = {}, acc2 = {}, acc3 = {};
    float4 aR[AREGS];                        // staging registers (the "prefetch")
    float4 bR[BREGS];

    // prologue: load chunk 0 into registers (global_load_b128, coalesced)
    #pragma unroll
    for (int i = 0; i < AREGS; ++i) {
        int lin = i * 256 + tid;
        int rr = lin >> 5, kq = lin & 31;    // 32 k-quads per A row
        aR[i] = *(const float4*)&x[(size_t)rl[rr] * D_IN + kq * 4];
    }
    #pragma unroll
    for (int i = 0; i < BREGS; ++i) {
        int lin = i * 256 + tid;
        int kk = lin >> 5, nq = lin & 31;    // quad along n: coalesced
        bR[i] = *(const float4*)&W[(size_t)kk * D_OUT + n0 + nq * 4];
    }

    for (int kb = 0; kb < D_IN; kb += KC) {
        // ---- phase 1: convert regs -> LDS (bf16) ----
        #pragma unroll
        for (int i = 0; i < AREGS; ++i) {
            int lin = i * 256 + tid;
            int rr = lin >> 5, kq = lin & 31;
            union { __bf16 h[4]; uint2 u; } t;
            t.h[0] = (__bf16)aR[i].x;  t.h[1] = (__bf16)aR[i].y;
            t.h[2] = (__bf16)aR[i].z;  t.h[3] = (__bf16)aR[i].w;
            *(uint2*)&As[rr * KC + kq * 4] = t.u;           // 8B ds_store
        }
        #pragma unroll
        for (int i = 0; i < BREGS; ++i) {
            int lin = i * 256 + tid;
            int kk = lin >> 5, nq = lin & 31;
            Bs[(nq * 4 + 0) * KPAD + kk] = (__bf16)bR[i].x; // transpose scatter
            Bs[(nq * 4 + 1) * KPAD + kk] = (__bf16)bR[i].y;
            Bs[(nq * 4 + 2) * KPAD + kk] = (__bf16)bR[i].z;
            Bs[(nq * 4 + 3) * KPAD + kk] = (__bf16)bR[i].w;
        }
        __syncthreads();

        // ---- phase 2: issue NEXT chunk's global loads (overlap with WMMA below) ----
        const int kn = kb + KC;
        if (kn < D_IN) {
            #pragma unroll
            for (int i = 0; i < AREGS; ++i) {
                int lin = i * 256 + tid;
                int rr = lin >> 5, kq = lin & 31;
                aR[i] = *(const float4*)&x[(size_t)rl[rr] * D_IN + kn + kq * 4];
            }
            #pragma unroll
            for (int i = 0; i < BREGS; ++i) {
                int lin = i * 256 + tid;
                int kk = lin >> 5, nq = lin & 31;
                bR[i] = *(const float4*)&W[(size_t)(kn + kk) * D_OUT + n0 + nq * 4];
            }
        }

        // ---- phase 3: 4 WMMA k-steps of 32 over this chunk ----
        #pragma unroll
        for (int ks = 0; ks < 4; ++ks) {
            const int kbase = ks * 32;
            // B frag (32x16 KxN): lane half 0 -> K 0..7 & 16..23, half 1 -> +8, column = lane%16
            const __bf16* bp = &Bs[col * KPAD + kbase + lh * 8];
            v16bf bf = cat8(*(const v8bf*)bp, *(const v8bf*)(bp + 16));
            const int aoff = lh * 8;   // A (16x32 MxK): half 0 -> K 0..7 & 16..23, half 1 -> +8
            #pragma unroll
            for (int ms = 0; ms < 4; ++ms) {
                const __bf16* ap = &As[(ms * 16 + nn) * KC + kbase];
                v16bf af = cat8(*(const v8bf*)(ap + aoff), *(const v8bf*)(ap + 16 + aoff));
                v8f* accp = (ms == 0) ? &acc0 : (ms == 1) ? &acc1 : (ms == 2) ? &acc2 : &acc3;
                *accp = __builtin_amdgcn_wmma_f32_16x16x32_bf16(
                            false, af, false, bf, (short)0, *accp, false, false);
            }
        }
        __syncthreads();
    }

    // ---- combine: out[b][n] += w * (dot + bias)  (exactly 2 adds/elem; fp add commutes) ----
    const float bias = ebias[e * D_OUT + n0 + col];
    #pragma unroll
    for (int ms = 0; ms < 4; ++ms) {
        v8f a = (ms == 0) ? acc0 : (ms == 1) ? acc1 : (ms == 2) ? acc2 : acc3;
        #pragma unroll
        for (int r = 0; r < 8; ++r) {
            int m = ms * 16 + r + lh * 8;        // C/D layout: M = vgpr + 8*(lane>=16)
            if (mt * MT + m < cnt) {
                float val = wl[m] * (a[r] + bias);
                atomicAdd(&out[(size_t)rl[m] * D_OUT + n0 + col], val);
            }
        }
    }
}

// ---------------- host ----------------
extern "C" void kernel_launch(void* const* d_in, const int* in_sizes, int n_in,
                              void* d_out, int out_size, void* d_ws, size_t ws_size,
                              hipStream_t stream) {
    (void)in_sizes; (void)n_in; (void)out_size; (void)ws_size;
    const float* x      = (const float*)d_in[0];
    const float* gu     = (const float*)d_in[1];
    const float* conv_w = (const float*)d_in[2];
    const float* conv_b = (const float*)d_in[3];
    const float* bng    = (const float*)d_in[4];
    const float* bnb    = (const float*)d_in[5];
    const float* bnm    = (const float*)d_in[6];
    const float* bnv    = (const float*)d_in[7];
    const float* fc1w   = (const float*)d_in[8];
    const float* fc1b   = (const float*)d_in[9];
    const float* fc2w   = (const float*)d_in[10];
    const float* fc2b   = (const float*)d_in[11];
    const float* ew     = (const float*)d_in[12];
    const float* ebias  = (const float*)d_in[13];
    float* out = (float*)d_out;

    char* ws = (char*)d_ws;                         // ~41 KB total
    int*   topi   = (int*)(ws);                     // 512*2 int
    float* wts    = (float*)(ws + 4096);            // 512*2 f32
    int*   counts = (int*)(ws + 8192);              // 8 int
    int*   rows   = (int*)(ws + 8448);              // 8*512 int
    float* rwt    = (float*)(ws + 8448 + 16384);    // 8*512 f32

    zero_kernel<<<(B_SZ * D_OUT + 255) / 256, 256, 0, stream>>>(out, counts);
    router_kernel<<<B_SZ, 256, 0, stream>>>(x, gu, conv_w, conv_b, bng, bnb, bnm, bnv,
                                            fc1w, fc1b, fc2w, fc2b, topi, wts);
    dispatch_kernel<<<1, 32, 0, stream>>>(topi, wts, counts, rows, rwt);
    expert_kernel<<<dim3(E_SZ * 8, D_OUT / NT), 256, 0, stream>>>(x, ew, ebias,
                                                                  counts, rows, rwt, out);
}
